// CustomImageCaptionModel_86818468921489
// MI455X (gfx1250) — matrix operand order repaired and checked by reading, not compiled, not run on hardware
//
#include <hip/hip_runtime.h>
#include <cstddef>

// ---------------------------------------------------------------------------
// MI455X (gfx1250) implementation of the image-caption model.
// All GEMM-shaped work (conv2-4 as implicit im2col GEMM, FC, LSTM gate GEMMs,
// vocab projection) runs on v_wmma_f32_16x16x32_bf16 (wave32). The vocab
// projection streams its bf16 weight tiles with the Tensor Data Mover
// (tensor_load_to_lds + s_wait_tensorcnt), with TDM padding producing the
// bank-conflict-free 36-element LDS row stride directly.
// ---------------------------------------------------------------------------

typedef __bf16    bf16x16 __attribute__((ext_vector_type(16)));
typedef float     f32x8   __attribute__((ext_vector_type(8)));
typedef unsigned  u32x4   __attribute__((ext_vector_type(4)));
typedef unsigned  u32x8   __attribute__((ext_vector_type(8)));

union Frag16 { bf16x16 v; unsigned u[8]; };

__device__ __forceinline__ unsigned short f32_to_bf16(float f) {
    union { float f; unsigned u; } x; x.f = f;
    unsigned r = x.u + 0x7FFFu + ((x.u >> 16) & 1u);   // round-to-nearest-even
    return (unsigned short)(r >> 16);
}
__device__ __forceinline__ float bf16_to_f32(unsigned short h) {
    union { unsigned u; float f; } x; x.u = ((unsigned)h) << 16;
    return x.f;
}
__device__ __forceinline__ float sigmoidf_(float x) { return 1.0f / (1.0f + __expf(-x)); }

// ---------------------------------------------------------------------------
// Layer 1: direct conv 3->32 + BN + ReLU + 2x2 maxpool (Cin=3 too small for
// WMMA; memory-bound on the 9.6MB input anyway). Output bf16 NCHW.
// ---------------------------------------------------------------------------
__global__ void conv1_bn_relu_pool(const float* __restrict__ img,
                                   const float* __restrict__ w,
                                   const float* __restrict__ bconv,
                                   const float* __restrict__ g,
                                   const float* __restrict__ be,
                                   const float* __restrict__ mu,
                                   const float* __restrict__ var,
                                   unsigned short* __restrict__ out) {
    const int C = 32, HO = 112, WO = 112, HI = 224, WI = 224;
    int idx = blockIdx.x * 256 + threadIdx.x;
    if (idx >= 16 * C * HO * WO) return;
    int px = idx % WO;
    int py = (idx / WO) % HO;
    int co = (idx / (WO * HO)) % C;
    int b  = idx / (WO * HO * C);
    float inv  = g[co] * rsqrtf(var[co] + 1e-5f);
    float beta = (bconv[co] - mu[co]) * inv + be[co];
    float best = -1e30f;
    #pragma unroll
    for (int dy = 0; dy < 2; ++dy) {
        #pragma unroll
        for (int dx = 0; dx < 2; ++dx) {
            int y = py * 2 + dy, x = px * 2 + dx;
            float acc = 0.f;
            #pragma unroll
            for (int ci = 0; ci < 3; ++ci) {
                #pragma unroll
                for (int ky = 0; ky < 3; ++ky) {
                    int yy = y + ky - 1;
                    if (yy < 0 || yy >= HI) continue;
                    #pragma unroll
                    for (int kx = 0; kx < 3; ++kx) {
                        int xx = x + kx - 1;
                        if (xx < 0 || xx >= WI) continue;
                        acc += img[((b * 3 + ci) * HI + yy) * WI + xx] *
                               w[((co * 3 + ci) * 3 + ky) * 3 + kx];
                    }
                }
            }
            float r = fmaxf(acc * inv + beta, 0.f);
            best = fmaxf(best, r);
        }
    }
    out[idx] = f32_to_bf16(best);
}

// ---------------------------------------------------------------------------
// Implicit-GEMM conv (3x3, pad 1, stride 1) with WMMA bf16.
//   C[M = B*H*W, N = Cout] = im2col(X)[M, K = Cin*9] x W[Cout, Cin*9]^T
// Fused BN+ReLU epilogue, bf16 NCHW output (pre-pool).
// Block: 256 threads = 8 waves; 64x64 tile, K step 32; wave = 16M x 32N.
// ---------------------------------------------------------------------------
__global__ __launch_bounds__(256)
void conv_wmma_bn_relu(const unsigned short* __restrict__ X,
                       const float* __restrict__ Wt,
                       const float* __restrict__ bconv,
                       const float* __restrict__ g,
                       const float* __restrict__ be,
                       const float* __restrict__ mu,
                       const float* __restrict__ var,
                       unsigned short* __restrict__ Y,
                       int Bn, int Cin, int H, int W, int Cout) {
    const int K = Cin * 9;
    const int Mtot = Bn * H * W;
    const int m0 = blockIdx.y * 64;
    const int n0 = blockIdx.x * 64;

    __shared__ __align__(16) unsigned short As[64][36];
    __shared__ __align__(16) unsigned short Bs[64][36];

    const int tid  = threadIdx.x;
    const int lane = tid & 31;
    const int wave = tid >> 5;
    const int wm   = wave & 3;   // 0..3 : M sub-tile
    const int wn   = wave >> 2;  // 0..1 : N sub-tile

    f32x8 acc0 = {}, acc1 = {};

    const int row = tid >> 2;          // 0..63
    const int cb  = (tid & 3) * 8;     // 0,8,16,24
    const int m   = m0 + row;
    const int xw  = m % W;
    const int yh  = (m / W) % H;
    const int bb  = m / (W * H);

    const int ktiles = K / 32;         // Cin*9 is a multiple of 32 for Cin in {32,64,128}
    for (int kt = 0; kt < ktiles; ++kt) {
        const int k0 = kt * 32;
        // ---- A tile: im2col gather, bf16 in / bf16 LDS ----
        #pragma unroll
        for (int j = 0; j < 8; ++j) {
            int k = k0 + cb + j;
            unsigned short val = 0;
            if (m < Mtot) {
                int ci = k / 9;
                int r9 = k - ci * 9;
                int ky = r9 / 3;
                int kx = r9 - ky * 3;
                int yy = yh + ky - 1, xx = xw + kx - 1;
                if (yy >= 0 && yy < H && xx >= 0 && xx < W)
                    val = X[((size_t)(bb * Cin + ci) * H + yy) * W + xx];
            }
            As[row][cb + j] = val;
        }
        // ---- B tile: weights fp32 [Cout][K] -> bf16 LDS ----
        {
            int n = n0 + row;
            if (n < Cout) {
                const float* src = Wt + (size_t)n * K + k0 + cb;
                if (kt + 1 < ktiles) __builtin_prefetch(src + 32, 0, 1);
                #pragma unroll
                for (int j = 0; j < 8; ++j) Bs[row][cb + j] = f32_to_bf16(src[j]);
            } else {
                #pragma unroll
                for (int j = 0; j < 8; ++j) Bs[row][cb + j] = 0;
            }
        }
        __syncthreads();

        // ---- fragments (ISA 7.12.2 16-bit layouts) ----
        const int fr = lane & 15;
        const int kh = lane >> 4;
        Frag16 a, bfr0, bfr1;
        #pragma unroll
        for (int p = 0; p < 8; ++p) {
            int kkA = 2 * p + ((p >= 4) ? 8 : 0) + 8 * kh;  // A: K-pair layout
            a.u[p] = *(const unsigned*)&As[wm * 16 + fr][kkA];
            int kkB = 2 * p + ((p >= 4) ? 8 : 0) + 8 * kh;  // B: K-pair layout (same as A)
            bfr0.u[p] = *(const unsigned*)&Bs[wn * 32 + fr][kkB];
            bfr1.u[p] = *(const unsigned*)&Bs[wn * 32 + 16 + fr][kkB];
        }
        acc0 = __builtin_amdgcn_wmma_f32_16x16x32_bf16(false, a.v, false, bfr0.v,
                                                       (short)0, acc0, false, false);
        acc1 = __builtin_amdgcn_wmma_f32_16x16x32_bf16(false, a.v, false, bfr1.v,
                                                       (short)0, acc1, false, false);
        __syncthreads();
    }

    // ---- epilogue: BN + ReLU, scatter to bf16 NCHW ----
    const int fr = lane & 15;
    const int hi = lane >> 4;
    #pragma unroll
    for (int tile = 0; tile < 2; ++tile) {
        int n = n0 + wn * 32 + tile * 16 + fr;
        if (n >= Cout) continue;
        float inv  = g[n] * rsqrtf(var[n] + 1e-5f);
        float beta = (bconv[n] - mu[n]) * inv + be[n];
        f32x8& acc = tile ? acc1 : acc0;
        #pragma unroll
        for (int p = 0; p < 8; ++p) {
            int mm = m0 + wm * 16 + p + 8 * hi;
            if (mm >= Mtot) continue;
            int ox = mm % W, oy = (mm / W) % H, ob = mm / (W * H);
            float val = fmaxf(acc[p] * inv + beta, 0.f);
            Y[((size_t)(ob * Cout + n) * H + oy) * W + ox] = f32_to_bf16(val);
        }
    }
}

// ---------------------------------------------------------------------------
// 2x2 max pool, bf16 -> bf16, NCHW.
// ---------------------------------------------------------------------------
__global__ void pool2x2(const unsigned short* __restrict__ in,
                        unsigned short* __restrict__ out,
                        int C, int H, int W, int total) {
    int idx = blockIdx.x * 256 + threadIdx.x;
    if (idx >= total) return;
    int WO = W / 2, HO = H / 2;
    int px = idx % WO;
    int py = (idx / WO) % HO;
    int c  = (idx / (WO * HO)) % C;
    int b  = idx / (WO * HO * C);
    const unsigned short* p = in + ((size_t)(b * C + c) * H + py * 2) * W + px * 2;
    float a0 = bf16_to_f32(p[0]),  a1 = bf16_to_f32(p[1]);
    float a2 = bf16_to_f32(p[W]),  a3 = bf16_to_f32(p[W + 1]);
    out[idx] = f32_to_bf16(fmaxf(fmaxf(a0, a1), fmaxf(a2, a3)));
}

// ---------------------------------------------------------------------------
// fp32 -> bf16 elementwise (one-time weight conversion for the TDM path).
// ---------------------------------------------------------------------------
__global__ void cvt_f32_bf16(const float* __restrict__ src,
                             unsigned short* __restrict__ dst, int n) {
    int i = blockIdx.x * 256 + threadIdx.x;
    if (i < n) dst[i] = f32_to_bf16(src[i]);
}

// ---------------------------------------------------------------------------
// Generic WMMA GEMM:  C[M,N] = act( A[M,K] * B + bias + bias2 (+addSrc) )
//   A: row-major, bf16 (ABF16) or fp32 (converted at tile load)
//   B: fp32, BNK ? row-major [N,K] : row-major [K,N]
//   addSrc (HASADD): fp32, element [m*addStride + n]
//   output: fp32 or bf16 (OUTBF16), row-major [M,N]
// Requires K % 32 == 0. Block 256 thr = 8 waves, 64x64 tile.
// ---------------------------------------------------------------------------
template <bool ABF16, bool BNK, bool RELU, bool OUTBF16, bool HASADD>
__global__ __launch_bounds__(256)
void gemm_wmma(const void* __restrict__ Aptr,
               const float* __restrict__ Bptr,
               const float* __restrict__ bias,
               const float* __restrict__ bias2,
               const float* __restrict__ addSrc, int addStride,
               void* __restrict__ Cptr,
               int M, int N, int K) {
    const int m0 = blockIdx.y * 64;
    const int n0 = blockIdx.x * 64;

    __shared__ __align__(16) unsigned short As[64][36];
    __shared__ __align__(16) unsigned short Bs[64][36];

    const int tid  = threadIdx.x;
    const int lane = tid & 31;
    const int wave = tid >> 5;
    const int wm   = wave & 3;
    const int wn   = wave >> 2;

    f32x8 acc0 = {}, acc1 = {};

    const unsigned short* Ab = (const unsigned short*)Aptr;
    const float*          Af = (const float*)Aptr;

    const int row = tid >> 2;
    const int cb  = (tid & 3) * 8;
    const int ktiles = K / 32;

    for (int kt = 0; kt < ktiles; ++kt) {
        const int k0 = kt * 32;
        // ---- A tile ----
        {
            int mrow = m0 + row;
            if (mrow < M) {
                if (ABF16) {
                    const unsigned short* src = Ab + (size_t)mrow * K + k0 + cb;
                    #pragma unroll
                    for (int j = 0; j < 8; ++j) As[row][cb + j] = src[j];
                } else {
                    const float* src = Af + (size_t)mrow * K + k0 + cb;
                    #pragma unroll
                    for (int j = 0; j < 8; ++j) As[row][cb + j] = f32_to_bf16(src[j]);
                }
            } else {
                #pragma unroll
                for (int j = 0; j < 8; ++j) As[row][cb + j] = 0;
            }
        }
        // ---- B tile (fp32 -> bf16) ----
        if (BNK) {
            int n = n0 + row;
            if (n < N) {
                const float* src = Bptr + (size_t)n * K + k0 + cb;
                if (kt + 1 < ktiles) __builtin_prefetch(src + 32, 0, 1);
                #pragma unroll
                for (int j = 0; j < 8; ++j) Bs[row][cb + j] = f32_to_bf16(src[j]);
            } else {
                #pragma unroll
                for (int j = 0; j < 8; ++j) Bs[row][cb + j] = 0;
            }
        } else {
            int n  = n0 + (tid & 63);
            int kb = (tid >> 6) * 8;
            if (n < N) {
                const float* src = Bptr + (size_t)(k0 + kb) * N + n;
                if (kt + 1 < ktiles) __builtin_prefetch(src + (size_t)32 * N, 0, 1);
                #pragma unroll
                for (int j = 0; j < 8; ++j)
                    Bs[tid & 63][kb + j] = f32_to_bf16(src[(size_t)j * N]);
            } else {
                #pragma unroll
                for (int j = 0; j < 8; ++j) Bs[tid & 63][kb + j] = 0;
            }
        }
        __syncthreads();

        const int fr = lane & 15;
        const int kh = lane >> 4;
        Frag16 a, bfr0, bfr1;
        #pragma unroll
        for (int p = 0; p < 8; ++p) {
            int kkA = 2 * p + ((p >= 4) ? 8 : 0) + 8 * kh;
            a.u[p] = *(const unsigned*)&As[wm * 16 + fr][kkA];
            int kkB = 2 * p + ((p >= 4) ? 8 : 0) + 8 * kh;
            bfr0.u[p] = *(const unsigned*)&Bs[wn * 32 + fr][kkB];
            bfr1.u[p] = *(const unsigned*)&Bs[wn * 32 + 16 + fr][kkB];
        }
        acc0 = __builtin_amdgcn_wmma_f32_16x16x32_bf16(false, a.v, false, bfr0.v,
                                                       (short)0, acc0, false, false);
        acc1 = __builtin_amdgcn_wmma_f32_16x16x32_bf16(false, a.v, false, bfr1.v,
                                                       (short)0, acc1, false, false);
        __syncthreads();
    }

    // ---- epilogue ----
    const int fr = lane & 15;
    const int hi = lane >> 4;
    #pragma unroll
    for (int tile = 0; tile < 2; ++tile) {
        int n = n0 + wn * 32 + tile * 16 + fr;
        if (n >= N) continue;
        float bsum = (bias ? bias[n] : 0.f) + (bias2 ? bias2[n] : 0.f);
        f32x8& acc = tile ? acc1 : acc0;
        #pragma unroll
        for (int p = 0; p < 8; ++p) {
            int mm = m0 + wm * 16 + p + 8 * hi;
            if (mm >= M) continue;
            float val = acc[p] + bsum;
            if (HASADD) val += addSrc[(size_t)mm * addStride + n];
            if (RELU)   val = fmaxf(val, 0.f);
            if (OUTBF16)
                ((unsigned short*)Cptr)[(size_t)mm * N + n] = f32_to_bf16(val);
            else
                ((float*)Cptr)[(size_t)mm * N + n] = val;
        }
    }
}

// ---------------------------------------------------------------------------
// WMMA GEMM with Tensor-Data-Mover weight streaming (vocab projection).
//   C[M,N] = A[M,K](fp32) * Bb[N,K](bf16)^T + bias
// Wave 0 DMAs the 32(K) x 64(N) bf16 weight tile into LDS via
// tensor_load_to_lds; TDM pad fields (interval=16 DW, amount=2 DW) produce
// the 36-element padded row stride directly. Tensor dims are given relative
// to the tile start so N-edge rows zero-fill in hardware.
// ---------------------------------------------------------------------------
__global__ __launch_bounds__(256)
void gemm_wmma_tdmB(const float* __restrict__ Af,
                    const unsigned short* __restrict__ Bb,
                    const float* __restrict__ bias,
                    float* __restrict__ Cf,
                    int M, int N, int K) {
    const int m0 = blockIdx.y * 64;
    const int n0 = blockIdx.x * 64;

    __shared__ __align__(16) unsigned short As[64][36];
    __shared__ __align__(16) unsigned short Bs[64][36];

    const int tid  = threadIdx.x;
    const int lane = tid & 31;
    const int wave = tid >> 5;
    const int wm   = wave & 3;
    const int wn   = wave >> 2;

    f32x8 acc0 = {}, acc1 = {};

    const int row = tid >> 2;
    const int cb  = (tid & 3) * 8;
    const int ktiles = K / 32;
    const unsigned lds_b = (unsigned)(size_t)(&Bs[0][0]);   // LDS byte offset (low 32b of aperture addr)

    for (int kt = 0; kt < ktiles; ++kt) {
        const int k0 = kt * 32;
        // ---- B tile via TDM (one wave issues; TDM ignores EXEC) ----
        if (wave == 0) {
            unsigned long long ga =
                (unsigned long long)(size_t)(Bb + (size_t)n0 * K + k0);
            unsigned td0 = (unsigned)(K - k0);   // remaining K (contiguous dim)
            unsigned td1 = (unsigned)(N - n0);   // remaining N rows (OOB rows -> 0)
            u32x4 g0;
            g0[0] = 1u;                                          // count=1, user desc
            g0[1] = lds_b;                                       // lds_addr
            g0[2] = (unsigned)(ga & 0xFFFFFFFFull);              // global_addr[31:0]
            g0[3] = (unsigned)((ga >> 32) & 0x01FFFFFFull)       // global_addr[56:32]
                    | (2u << 30);                                // type=2 (image)
            u32x8 g1;
            g1[0] = (1u << 16)        // data_size = 2 bytes
                  | (1u << 20)        // pad_enable
                  | (3u << 22)        // pad_interval: 16 DWORDs (= 32 bf16 row)
                  | (1u << 25);       // pad_amount: 2 DWORDs (= 4 bf16 -> stride 36)
            g1[1] = (td0 & 0xFFFFu) << 16;                       // tensor_dim0[15:0]
            g1[2] = (td0 >> 16) | ((td1 & 0xFFFFu) << 16);       // dim0 hi | dim1 lo
            g1[3] = (td1 >> 16) | (32u << 16);                   // dim1 hi | tile_dim0=32
            g1[4] = 64u;                                         // tile_dim1=64, tile_dim2=0
            g1[5] = (unsigned)K;                                 // tensor_dim0_stride lo
            g1[6] = 0u;                                          // stride hi | dim1_stride lo
            g1[7] = 0u;
            asm volatile("tensor_load_to_lds %0, %1" : : "s"(g0), "s"(g1) : "memory");
            __builtin_amdgcn_s_wait_tensorcnt(0);
        }
        // ---- A tile: fp32 -> bf16 (overlaps with TDM DMA) ----
        {
            int mrow = m0 + row;
            if (mrow < M) {
                const float* src = Af + (size_t)mrow * K + k0 + cb;
                if (kt + 1 < ktiles) __builtin_prefetch(src + 32, 0, 1);
                #pragma unroll
                for (int j = 0; j < 8; ++j) As[row][cb + j] = f32_to_bf16(src[j]);
            } else {
                #pragma unroll
                for (int j = 0; j < 8; ++j) As[row][cb + j] = 0;
            }
        }
        __syncthreads();

        const int fr = lane & 15;
        const int kh = lane >> 4;
        Frag16 a, bfr0, bfr1;
        #pragma unroll
        for (int p = 0; p < 8; ++p) {
            int kkA = 2 * p + ((p >= 4) ? 8 : 0) + 8 * kh;
            a.u[p] = *(const unsigned*)&As[wm * 16 + fr][kkA];
            int kkB = 2 * p + ((p >= 4) ? 8 : 0) + 8 * kh;
            bfr0.u[p] = *(const unsigned*)&Bs[wn * 32 + fr][kkB];
            bfr1.u[p] = *(const unsigned*)&Bs[wn * 32 + 16 + fr][kkB];
        }
        acc0 = __builtin_amdgcn_wmma_f32_16x16x32_bf16(false, a.v, false, bfr0.v,
                                                       (short)0, acc0, false, false);
        acc1 = __builtin_amdgcn_wmma_f32_16x16x32_bf16(false, a.v, false, bfr1.v,
                                                       (short)0, acc1, false, false);
        __syncthreads();
    }

    // ---- epilogue ----
    const int fr = lane & 15;
    const int hi = lane >> 4;
    #pragma unroll
    for (int tile = 0; tile < 2; ++tile) {
        int n = n0 + wn * 32 + tile * 16 + fr;
        if (n >= N) continue;
        float bsum = bias ? bias[n] : 0.f;
        f32x8& acc = tile ? acc1 : acc0;
        #pragma unroll
        for (int p = 0; p < 8; ++p) {
            int mm = m0 + wm * 16 + p + 8 * hi;
            if (mm >= M) continue;
            Cf[(size_t)mm * N + n] = acc[p] + bsum;
        }
    }
}

// ---------------------------------------------------------------------------
// Build LSTM input sequence [B=16, T=24, 3E=1536] in bf16:
//   [img_feat(512) | emo_emb[emotions[b]](512) | tok_emb[captions[b,t]](512)]
// ---------------------------------------------------------------------------
__global__ void assemble_inputs(const unsigned short* __restrict__ img_feat,
                                const float* __restrict__ emo_emb,
                                const float* __restrict__ tok_emb,
                                const int* __restrict__ captions,
                                const int* __restrict__ emotions,
                                unsigned short* __restrict__ inp) {
    int idx = blockIdx.x * 256 + threadIdx.x;
    if (idx >= 384 * 1536) return;
    int e = idx % 1536;
    int t = (idx / 1536) % 24;
    int b = idx / (1536 * 24);
    unsigned short val;
    if (e < 512)       val = img_feat[b * 512 + e];
    else if (e < 1024) val = f32_to_bf16(emo_emb[(size_t)emotions[b] * 512 + (e - 512)]);
    else               val = f32_to_bf16(tok_emb[(size_t)captions[b * 24 + t] * 512 + (e - 1024)]);
    inp[idx] = val;
}

// ---------------------------------------------------------------------------
// LSTM cell pointwise update. gates [16,4096] fp32 (i|f|g|o), h/c [16,1024].
// ---------------------------------------------------------------------------
__global__ void lstm_cell(const float* __restrict__ gates,
                          float* __restrict__ c,
                          float* __restrict__ h,
                          float* __restrict__ lstm_out, int t) {
    int idx = blockIdx.x * 256 + threadIdx.x;
    if (idx >= 16 * 1024) return;
    int j = idx % 1024, b = idx / 1024;
    const float* gr = gates + (size_t)b * 4096;
    float ig = sigmoidf_(gr[j]);
    float fg = sigmoidf_(gr[1024 + j]);
    float gg = tanhf(gr[2048 + j]);
    float og = sigmoidf_(gr[3072 + j]);
    float cn = fg * c[idx] + ig * gg;
    float hn = og * tanhf(cn);
    c[idx] = cn;
    h[idx] = hn;
    lstm_out[((size_t)b * 24 + t) * 1024 + j] = hn;
}

// ---------------------------------------------------------------------------
// Host launcher
// ---------------------------------------------------------------------------
static inline unsigned ceil_div(unsigned a, unsigned b) { return (a + b - 1) / b; }

extern "C" void kernel_launch(void* const* d_in, const int* in_sizes, int n_in,
                              void* d_out, int out_size, void* d_ws, size_t ws_size,
                              hipStream_t stream) {
    (void)in_sizes; (void)n_in; (void)out_size; (void)ws_size;

    const float* images   = (const float*)d_in[0];
    const int*   captions = (const int*)d_in[1];
    const int*   emotions = (const int*)d_in[2];
    const float *w[4], *bc[4], *g[4], *be[4], *mu[4], *vv[4];
    for (int i = 0; i < 4; ++i) {
        w[i]  = (const float*)d_in[3 + i * 6 + 0];
        bc[i] = (const float*)d_in[3 + i * 6 + 1];
        g[i]  = (const float*)d_in[3 + i * 6 + 2];
        be[i] = (const float*)d_in[3 + i * 6 + 3];
        mu[i] = (const float*)d_in[3 + i * 6 + 4];
        vv[i] = (const float*)d_in[3 + i * 6 + 5];
    }
    const float* fc_w    = (const float*)d_in[27];
    const float* fc_b    = (const float*)d_in[28];
    const float* emo_emb = (const float*)d_in[29];
    const float* tok_emb = (const float*)d_in[30];
    const float* W_ih    = (const float*)d_in[31];
    const float* W_hh    = (const float*)d_in[32];
    const float* b_ih    = (const float*)d_in[33];
    const float* b_hh    = (const float*)d_in[34];
    const float* out_w   = (const float*)d_in[35];
    const float* out_b   = (const float*)d_in[36];

    char* ws = (char*)d_ws;
    // ping-pong bf16 activation buffers; OFF_B region is reused for the bf16
    // copy of out_w once the conv stages are done (needs 20.48MB < 27MB gap).
    const size_t OFF_A   = 0;                       // <= 12.85 MB (pool outputs)
    const size_t OFF_B   = (size_t)16 << 20;        // <= 25.7 MB (conv outputs)
    const size_t OFF_IMG = (size_t)43 << 20;        // img_feat bf16 [16,512]
    const size_t OFF_INP = OFF_IMG + ((size_t)1 << 20);   // inputs bf16 [384,1536]
    const size_t OFF_XP  = OFF_INP + ((size_t)2 << 20);   // XP fp32 [384,4096]
    const size_t OFF_GT  = OFF_XP  + ((size_t)7 << 20);   // gates fp32 [16,4096]
    const size_t OFF_H   = OFF_GT  + ((size_t)1 << 20);   // h fp32 [16,1024]
    const size_t OFF_C   = OFF_H   + ((size_t)1 << 20);   // c fp32 [16,1024]
    const size_t OFF_LO  = OFF_C   + ((size_t)1 << 20);   // lstm_out fp32 [384,1024]

    unsigned short* bufA     = (unsigned short*)(ws + OFF_A);
    unsigned short* bufB     = (unsigned short*)(ws + OFF_B);
    unsigned short* out_wb   = (unsigned short*)(ws + OFF_B);   // aliases bufB (post-conv)
    unsigned short* img_feat = (unsigned short*)(ws + OFF_IMG);
    unsigned short* inputs   = (unsigned short*)(ws + OFF_INP);
    float* XP       = (float*)(ws + OFF_XP);
    float* gates    = (float*)(ws + OFF_GT);
    float* hbuf     = (float*)(ws + OFF_H);
    float* cbuf     = (float*)(ws + OFF_C);
    float* lstm_out = (float*)(ws + OFF_LO);
    float* logits   = (float*)d_out;

    // ---- Stage 1: conv1 (direct, fused pool) -> bufA [16,32,112,112] bf16 ----
    conv1_bn_relu_pool<<<ceil_div(16u * 32 * 112 * 112, 256), 256, 0, stream>>>(
        images, w[0], bc[0], g[0], be[0], mu[0], vv[0], bufA);

    // ---- conv2 (WMMA): bufA -> bufB [16,64,112,112]; pool -> bufA [16,64,56,56] ----
    conv_wmma_bn_relu<<<dim3(1, 16 * 112 * 112 / 64), 256, 0, stream>>>(
        bufA, w[1], bc[1], g[1], be[1], mu[1], vv[1], bufB, 16, 32, 112, 112, 64);
    pool2x2<<<ceil_div(16u * 64 * 56 * 56, 256), 256, 0, stream>>>(
        bufB, bufA, 64, 112, 112, 16 * 64 * 56 * 56);

    // ---- conv3: bufA -> bufB [16,128,56,56]; pool -> bufA [16,128,28,28] ----
    conv_wmma_bn_relu<<<dim3(2, 16 * 56 * 56 / 64), 256, 0, stream>>>(
        bufA, w[2], bc[2], g[2], be[2], mu[2], vv[2], bufB, 16, 64, 56, 56, 128);
    pool2x2<<<ceil_div(16u * 128 * 28 * 28, 256), 256, 0, stream>>>(
        bufB, bufA, 128, 56, 56, 16 * 128 * 28 * 28);

    // ---- conv4: bufA -> bufB [16,256,28,28]; pool -> bufA [16,256,14,14] ----
    conv_wmma_bn_relu<<<dim3(4, 16 * 28 * 28 / 64), 256, 0, stream>>>(
        bufA, w[3], bc[3], g[3], be[3], mu[3], vv[3], bufB, 16, 128, 28, 28, 256);
    pool2x2<<<ceil_div(16u * 256 * 14 * 14, 256), 256, 0, stream>>>(
        bufB, bufA, 256, 28, 28, 16 * 256 * 14 * 14);

    // ---- out_w -> bf16 (bufB region now dead; enables TDM weight streaming) ----
    cvt_f32_bf16<<<ceil_div(10000u * 1024, 256), 256, 0, stream>>>(
        out_w, out_wb, 10000 * 1024);

    // ---- FC: img_feat = relu(x @ fc_w^T + fc_b), bf16 out [16,512] ----
    gemm_wmma<true, true, true, true, false><<<dim3(512 / 64, 1), 256, 0, stream>>>(
        bufA, fc_w, fc_b, nullptr, nullptr, 0, img_feat, 16, 512, 50176);

    // ---- assemble LSTM inputs [384,1536] bf16 ----
    assemble_inputs<<<ceil_div(384u * 1536, 256), 256, 0, stream>>>(
        img_feat, emo_emb, tok_emb, captions, emotions, inputs);

    // ---- XP = inputs @ W_ih + b_ih + b_hh : [384,4096] fp32 (B is [K,N]) ----
    gemm_wmma<true, false, false, false, false><<<dim3(4096 / 64, 384 / 64), 256, 0, stream>>>(
        inputs, W_ih, b_ih, b_hh, nullptr, 0, XP, 384, 4096, 1536);

    // ---- LSTM recurrence ----
    hipMemsetAsync(hbuf, 0, 16 * 1024 * sizeof(float), stream);
    hipMemsetAsync(cbuf, 0, 16 * 1024 * sizeof(float), stream);
    for (int t = 0; t < 24; ++t) {
        gemm_wmma<false, false, false, false, true><<<dim3(4096 / 64, 1), 256, 0, stream>>>(
            hbuf, W_hh, nullptr, nullptr, XP + (size_t)t * 4096, 24 * 4096,
            gates, 16, 4096, 1024);
        lstm_cell<<<ceil_div(16u * 1024, 256), 256, 0, stream>>>(
            gates, cbuf, hbuf, lstm_out, t);
    }

    // ---- vocab projection via TDM weight streaming:
    //      logits = lstm_out @ out_wb^T + out_b : [384,10000] fp32 ----
    gemm_wmma_tdmB<<<dim3(ceil_div(10000, 64), 384 / 64), 256, 0, stream>>>(
        lstm_out, out_wb, out_b, logits, 384, 10000, 1024);
}
